// SimpleGNN_28338194219589
// MI455X (gfx1250) — matrix operand order repaired and checked by reading, not compiled, not run on hardware
//
#include <hip/hip_runtime.h>

#define NNODES 100000
#define NEDGES 1600000
#define FIN 128
#define H1F 64
#define H2F 32

typedef __attribute__((ext_vector_type(16))) __bf16 v16bf;
typedef __attribute__((ext_vector_type(8)))  float  v8f;

// ---------------- utility kernels ----------------

__global__ __launch_bounds__(256) void k_fill(float* __restrict__ p, float v, int n) {
  int i = blockIdx.x * blockDim.x + threadIdx.x;
  if (i < n) p[i] = v;
}

__global__ __launch_bounds__(256) void k_deg(const int* __restrict__ dst,
                                             float* __restrict__ deg, int ne) {
  int i = blockIdx.x * blockDim.x + threadIdx.x;
  if (i < ne) atomicAdd(&deg[dst[i]], 1.0f);
}

__global__ __launch_bounds__(256) void k_rsqrt_inplace(float* __restrict__ d, int n) {
  int i = blockIdx.x * blockDim.x + threadIdx.x;
  if (i < n) d[i] = rsqrtf(d[i]);   // deg >= 1 always (self-loops)
}

__global__ __launch_bounds__(256) void k_relu(float* __restrict__ a, int n) {
  int i = blockIdx.x * blockDim.x + threadIdx.x;
  if (i < n) a[i] = fmaxf(a[i], 0.0f);
}

// agg = dinv[node]^2 * h + bias   (folds the self-loop term and the bias)
template <int H>
__global__ __launch_bounds__(256) void k_prime(const float* __restrict__ h,
                                               const float* __restrict__ dinv,
                                               const float* __restrict__ bias,
                                               float* __restrict__ agg, int n) {
  int idx = blockIdx.x * blockDim.x + threadIdx.x;
  if (idx < n * H) {
    int node = idx / H;
    int f    = idx - node * H;
    float di = dinv[node];
    agg[idx] = di * di * h[idx] + bias[f];
  }
}

// ---------------- WMMA bf16 GEMM: C[M x NOUT] = A[M x K] @ W[K x NOUT] ----------------
// W is pre-packed once per block into LDS in this kernel's contiguous-k bf16 B-fragment
// layout (same k order as A): per (k-tile, n-tile), lane = (k%32>=16)*16 + col holds the
// 16 K-consecutive bf16 values as one contiguous 32B vector -> single ds_load.
// Each wave computes a 16 x NOUT strip: the A fragment (converted f32->bf16 on
// the fly) is loaded once per K-step and reused across all N-tiles.
template <int K, int NOUT>
__global__ __launch_bounds__(256) void gemm_bf16_wmma(const float* __restrict__ A,
                                                      const float* __restrict__ W,
                                                      float* __restrict__ C, int M) {
  constexpr int TILES_N = NOUT / 16;
  constexpr int KTILES  = K / 32;
  __shared__ __align__(32) __bf16 ldsW[KTILES * TILES_N * 32 * 16];  // = K*NOUT bf16

  // cooperative pack of W into B-fragment layout (all 256 threads, before any exit)
  for (int i = threadIdx.x; i < K * NOUT; i += 256) {
    int k = i / NOUT, n = i - (i / NOUT) * NOUT;
    int kt = k >> 5, kr = k & 31;
    int nt = n >> 4, col = n & 15;
    int lane = ((kr >> 4) << 4) + col;
    int e    = kr & 15;
    ldsW[(((kt * TILES_N + nt) * 32 + lane) << 4) + e] = (__bf16)W[i];
  }
  __syncthreads();

  const int lane = threadIdx.x & 31;
  const int wave = blockIdx.x * 8 + (threadIdx.x >> 5);
  if (wave >= M / 16) return;                 // wave-uniform, after the barrier

  const int idx15 = lane & 15;                // M (A rows) / N (B/C cols) index
  const int half  = lane >> 4;                // K-half selector
  const int m0    = wave * 16;

  v8f acc[TILES_N];
#pragma unroll
  for (int nt = 0; nt < TILES_N; ++nt) acc[nt] = {};

#pragma unroll
  for (int kt = 0; kt < KTILES; ++kt) {
    // A 16x32 bf16 fragment: lane holds row idx15, K = half*16 + {0..15}
    v16bf a;
    const float* arow = A + (size_t)(m0 + idx15) * K + kt * 32 + half * 16;
#pragma unroll
    for (int j = 0; j < 8; ++j) {
      a[j]     = (__bf16)arow[j];
      a[j + 8] = (__bf16)arow[j + 8];
    }
#pragma unroll
    for (int nt = 0; nt < TILES_N; ++nt) {
      v16bf b = *(const v16bf*)&ldsW[((kt * TILES_N + nt) * 32 + lane) << 4];
      acc[nt] = __builtin_amdgcn_wmma_f32_16x16x32_bf16(
          /*neg_a=*/false, a, /*neg_b=*/false, b,
          /*c_mod=*/(short)0, acc[nt], /*reuse_a=*/false, /*reuse_b=*/false);
    }
  }
  // C/D f32 16x16: VGPR r -> row r + half*8, lane -> column idx15
#pragma unroll
  for (int nt = 0; nt < TILES_N; ++nt)
#pragma unroll
    for (int r = 0; r < 8; ++r)
      C[(size_t)(m0 + r + half * 8) * NOUT + nt * 16 + idx15] = acc[nt][r];
}

// ---------------- edge scatter: agg[dst] += dinv[src]*dinv[dst]*h[src] ----------------
// One wave per edge; features striped across lanes. The per-node feature buffers
// (<=25.6MB) stay resident in the 192MB L2, so gathers and atomic-RMWs resolve at
// L2 (which hosts the atomic units), not HBM.
template <int H>
__global__ __launch_bounds__(256) void k_scatter(const int* __restrict__ src,
                                                 const int* __restrict__ dst,
                                                 const float* __restrict__ dinv,
                                                 const float* __restrict__ h,
                                                 float* __restrict__ agg, int ne) {
  const int lane = threadIdx.x & 31;
  const int e = (blockIdx.x * blockDim.x + threadIdx.x) >> 5;
  if (e >= ne) return;
  const int s = src[e];
  const int d = dst[e];
  const float w = dinv[s] * dinv[d];
  if (H == 64) {
    const int f = lane * 2;
    const float2 v = *(const float2*)(h + (size_t)s * 64 + f);
    atomicAdd(agg + (size_t)d * 64 + f,     w * v.x);
    atomicAdd(agg + (size_t)d * 64 + f + 1, w * v.y);
  } else {
    const float v = h[(size_t)s * 32 + lane];
    atomicAdd(agg + (size_t)d * 32 + lane, w * v);
  }
}

// ---------------- head: out[i] = dot(a2[i,:32], Wfc) + bfc ----------------
__global__ __launch_bounds__(256) void k_final(const float* __restrict__ a2,
                                               const float* __restrict__ Wfc,
                                               const float* __restrict__ bfc,
                                               float* __restrict__ out, int n) {
  const int lane = threadIdx.x & 31;
  const int node = (blockIdx.x * blockDim.x + threadIdx.x) >> 5;
  if (node >= n) return;
  float v = a2[(size_t)node * 32 + lane] * Wfc[lane];
#pragma unroll
  for (int m = 16; m >= 1; m >>= 1) v += __shfl_xor(v, m, 32);
  if (lane == 0) out[node] = v + bfc[0];
}

// ---------------- launch ----------------

extern "C" void kernel_launch(void* const* d_in, const int* in_sizes, int n_in,
                              void* d_out, int out_size, void* d_ws, size_t ws_size,
                              hipStream_t stream) {
  const float* x   = (const float*)d_in[0];
  const int*   ei  = (const int*)d_in[1];
  const float* W1  = (const float*)d_in[2];
  const float* b1  = (const float*)d_in[3];
  const float* W2  = (const float*)d_in[4];
  const float* b2  = (const float*)d_in[5];
  const float* Wfc = (const float*)d_in[6];
  const float* bfc = (const float*)d_in[7];
  float* out = (float*)d_out;

  const int* src = ei;            // edge_index[0]
  const int* dst = ei + NEDGES;   // edge_index[1]

  float* ws   = (float*)d_ws;
  float* dinv = ws;                          // N   (deg, then rsqrt in place)
  float* h1   = dinv + NNODES;               // N*64
  float* agg1 = h1 + (size_t)NNODES * H1F;   // N*64
  float* h2   = agg1 + (size_t)NNODES * H1F; // N*32
  float* agg2 = h2 + (size_t)NNODES * H2F;   // N*32

  const int B = 256;
  auto cdiv = [](long long a, long long b) { return (int)((a + b - 1) / b); };

  // degrees (self-loop baked in as deg=1), then dinv = rsqrt(deg)
  k_fill<<<cdiv(NNODES, B), B, 0, stream>>>(dinv, 1.0f, NNODES);
  k_deg<<<cdiv(NEDGES, B), B, 0, stream>>>(dst, dinv, NEDGES);
  k_rsqrt_inplace<<<cdiv(NNODES, B), B, 0, stream>>>(dinv, NNODES);

  // layer 1: h1 = X @ W1 (WMMA), agg1 = dinv^2*h1 + b1, scatter edges, relu
  gemm_bf16_wmma<FIN, H1F><<<cdiv(NNODES / 16, 8), B, 0, stream>>>(x, W1, h1, NNODES);
  k_prime<H1F><<<cdiv((long long)NNODES * H1F, B), B, 0, stream>>>(h1, dinv, b1, agg1, NNODES);
  k_scatter<H1F><<<cdiv((long long)NEDGES * 32, B), B, 0, stream>>>(src, dst, dinv, h1, agg1, NEDGES);
  k_relu<<<cdiv((long long)NNODES * H1F, B), B, 0, stream>>>(agg1, NNODES * H1F);

  // layer 2: h2 = agg1 @ W2 (WMMA), agg2 = dinv^2*h2 + b2, scatter edges, relu
  gemm_bf16_wmma<H1F, H2F><<<cdiv(NNODES / 16, 8), B, 0, stream>>>(agg1, W2, h2, NNODES);
  k_prime<H2F><<<cdiv((long long)NNODES * H2F, B), B, 0, stream>>>(h2, dinv, b2, agg2, NNODES);
  k_scatter<H2F><<<cdiv((long long)NEDGES * 32, B), B, 0, stream>>>(src, dst, dinv, h2, agg2, NEDGES);
  k_relu<<<cdiv((long long)NNODES * H2F, B), B, 0, stream>>>(agg2, NNODES * H2F);

  // head: out = agg2 @ Wfc + bfc (wave32 shuffle reduction)
  k_final<<<cdiv((long long)NNODES * 32, B), B, 0, stream>>>(agg2, Wfc, bfc, out, NNODES);
}
